// GeneratedTreeClassifier_57758720197280
// MI455X (gfx1250) — matrix operand order repaired and checked by reading, not compiled, not run on hardware
//
#include <hip/hip_runtime.h>

typedef __attribute__((ext_vector_type(16))) __bf16 bf16x16;
typedef __attribute__((ext_vector_type(8)))  __bf16 bf16x8;
typedef __attribute__((ext_vector_type(8)))  float  f32x8;

#define B_SZ   4096   // batch
#define D_SZ   512    // features
#define T_SZ   64     // trees
#define N_NODES 63    // internal nodes per tree
#define L_SZ   64     // leaves per tree
#define C_SZ   100    // classes
#define C_PAD  112    // classes padded to multiple of 16
#define K2     4096   // T_SZ * L_SZ  (K of GEMM2, also padded node columns of GEMM1)

__device__ __forceinline__ __bf16 f2bf(float f) {
  unsigned u = __builtin_bit_cast(unsigned, f);
  unsigned r = u + 0x7FFFu + ((u >> 16) & 1u);   // round-to-nearest-even
  unsigned short h = (unsigned short)(r >> 16);
  return __builtin_bit_cast(__bf16, h);
}

// ---- prep: x fp32 -> bf16 (row-major [B][D]) -------------------------------
__global__ void cvt_x_kernel(const float* __restrict__ x, __bf16* __restrict__ xb) {
  int i = blockIdx.x * blockDim.x + threadIdx.x;
  xb[i] = f2bf(x[i]);
}

// ---- prep: split_weights (T,63,D) -> padded column-major-for-GEMM bf16 -----
// wp[(t*64+n)*D + d]; row n==63 is zero padding.
__global__ void pack_w_kernel(const float* __restrict__ sw, __bf16* __restrict__ wp) {
  int i = blockIdx.x * blockDim.x + threadIdx.x;   // over K2*D
  int c = i >> 9;          // /D_SZ
  int d = i & (D_SZ - 1);
  int t = c >> 6;
  int n = c & 63;
  float v = (n < N_NODES) ? sw[(t * N_NODES + n) * D_SZ + d] : 0.0f;
  wp[i] = f2bf(v);
}

// ---- prep: dist_col[c][t*64+l] = softmax(tree_w)[t] * softmax(leaf_logits[t,l,:])[c]
__global__ void dist_kernel(const float* __restrict__ leaf_logits,
                            const float* __restrict__ tree_w,
                            __bf16* __restrict__ dist_col) {
  int row = blockIdx.x * blockDim.x + threadIdx.x;   // 0..K2-1, row = t*64+l
  if (row >= K2) return;
  int t = row >> 6;

  float mx = tree_w[0];
  for (int i = 1; i < T_SZ; ++i) mx = fmaxf(mx, tree_w[i]);
  float se = 0.f;
  for (int i = 0; i < T_SZ; ++i) se += __expf(tree_w[i] - mx);
  float w = __expf(tree_w[t] - mx) / se;

  const float* ll = leaf_logits + row * C_SZ;
  float m2 = ll[0];
  for (int c = 1; c < C_SZ; ++c) m2 = fmaxf(m2, ll[c]);
  float s2 = 0.f;
  for (int c = 0; c < C_SZ; ++c) s2 += __expf(ll[c] - m2);
  float inv = w / s2;
  for (int c = 0; c < C_SZ; ++c)
    dist_col[(size_t)c * K2 + row] = f2bf(__expf(ll[c] - m2) * inv);
  for (int c = C_SZ; c < C_PAD; ++c)
    dist_col[(size_t)c * K2 + row] = f2bf(0.0f);
}

// ---- GEMM1 (bf16 WMMA) fused with sigmoid + leaf-path products -------------
// Each wave: one 16(M) x 64(N) tile == 16 batch rows x one whole tree t.
// Block = 8 waves = one M-tile x 8 trees.
__global__ __launch_bounds__(256) void gemm1_leafprob_kernel(
    const __bf16* __restrict__ xb,     // [B][D] bf16
    const __bf16* __restrict__ wp,     // [K2][D] bf16 (K-contiguous per column)
    const float*  __restrict__ bias,   // [T][63] fp32
    __bf16*       __restrict__ a2)     // out: [B][K2] bf16 leaf probs
{
  __shared__ float sT[8][16 * 64];     // per-wave sigmoid tile (32 KB total)

  int lane = threadIdx.x & 31;
  int wid  = threadIdx.x >> 5;
  int gw   = blockIdx.x * 8 + wid;
  int t    = gw & 63;                  // tree index
  int mt   = gw >> 6;                  // M-tile (16 rows of x)
  int mbase = mt * 16;

  // WMMA bf16 fragment addressing (wave32, contiguous k order: k = 16*(lane>>4) + s in A and B)
  int am  = lane & 15;                 // A: row within tile
  int akb = (lane >> 4) * 16;          // A: K sub-block base
  int bn  = lane & 15;                 // B: column within 16
  int bkb = (lane >> 4) * 16;          // B: K sub-block base

  f32x8 acc[4];
#pragma unroll
  for (int j = 0; j < 4; ++j) acc[j] = (f32x8){};

  const __bf16* arow  = xb + (size_t)(mbase + am) * D_SZ;
  const __bf16* bbase = wp + (size_t)(t * 64) * D_SZ;

  for (int k0 = 0; k0 < D_SZ; k0 += 32) {
    bf16x8 alo = *(const bf16x8*)(arow + k0 + akb);
    bf16x8 ahi = *(const bf16x8*)(arow + k0 + 8 + akb);
    bf16x16 a;
#pragma unroll
    for (int i = 0; i < 8; ++i) { a[i] = alo[i]; a[i + 8] = ahi[i]; }
#pragma unroll
    for (int j = 0; j < 4; ++j) {
      bf16x16 b = *(const bf16x16*)(bbase + (size_t)(bn + 16 * j) * D_SZ + k0 + bkb);
      acc[j] = __builtin_amdgcn_wmma_f32_16x16x32_bf16(
                  false, a, false, b, (short)0, acc[j], false, false);
    }
  }

  // epilogue: add bias, sigmoid, scatter node probs to LDS tile
  int mrow0 = (lane >> 4) * 8;         // C/D layout: lane holds rows mrow0..mrow0+7
  float* swt = sT[wid];
#pragma unroll
  for (int j = 0; j < 4; ++j) {
    int n = (lane & 15) + 16 * j;      // node slot 0..63 within tree
    float bv = (n < N_NODES) ? bias[t * N_NODES + n] : 0.0f;
#pragma unroll
    for (int v = 0; v < 8; ++v) {
      float lg = acc[j][v] + bv;
      float s  = 1.0f / (1.0f + __expf(-lg));
      swt[(mrow0 + v) * 64 + n] = s;
    }
  }
  __syncthreads();

  // leaf probabilities: lane -> (row r, leaf half h); 32 leaves each, depth-6 path
  int r = lane & 15;
  int h = lane >> 4;
  const float* srow = swt + r * 64;
  __bf16* orow = a2 + (size_t)(mbase + r) * K2 + t * 64 + h * 32;
#pragma unroll 4
  for (int j = 0; j < 32; ++j) {
    int lf = h * 32 + j;
    float p = 2.0f;                    // the reference's 2.0 * prod
    int node = 0;
#pragma unroll
    for (int k = 5; k >= 0; --k) {
      int bit = (lf >> k) & 1;
      float s = srow[node];
      p *= bit ? s : (1.0f - s);
      node = 2 * node + 1 + bit;
    }
    orow[j] = f2bf(p);
  }
}

// ---- GEMM2 (bf16 WMMA): out[B][100] = A2[B][K2] @ dist_col^T ----------------
// Each wave: 16(M) x 112(N=7 tiles), K = 4096.
__global__ __launch_bounds__(256) void gemm2_kernel(
    const __bf16* __restrict__ a2,       // [B][K2]
    const __bf16* __restrict__ dist,     // [C_PAD][K2] (K-contiguous per class)
    float*        __restrict__ out)      // [B][C_SZ]
{
  int lane = threadIdx.x & 31;
  int wid  = threadIdx.x >> 5;
  int mt   = blockIdx.x * 8 + wid;       // 0..255
  int mbase = mt * 16;

  int am  = lane & 15;
  int akb = (lane >> 4) * 16;
  int bn  = lane & 15;
  int bkb = (lane >> 4) * 16;

  f32x8 acc[7];
#pragma unroll
  for (int j = 0; j < 7; ++j) acc[j] = (f32x8){};

  const __bf16* arow = a2 + (size_t)(mbase + am) * K2;

  for (int k0 = 0; k0 < K2; k0 += 32) {
    bf16x8 alo = *(const bf16x8*)(arow + k0 + akb);
    bf16x8 ahi = *(const bf16x8*)(arow + k0 + 8 + akb);
    bf16x16 a;
#pragma unroll
    for (int i = 0; i < 8; ++i) { a[i] = alo[i]; a[i + 8] = ahi[i]; }
#pragma unroll
    for (int j = 0; j < 7; ++j) {
      bf16x16 b = *(const bf16x16*)(dist + (size_t)(bn + 16 * j) * K2 + k0 + bkb);
      acc[j] = __builtin_amdgcn_wmma_f32_16x16x32_bf16(
                  false, a, false, b, (short)0, acc[j], false, false);
    }
  }

  int mrow0 = (lane >> 4) * 8;
#pragma unroll
  for (int j = 0; j < 7; ++j) {
    int c = (lane & 15) + 16 * j;
    if (c < C_SZ) {
#pragma unroll
      for (int v = 0; v < 8; ++v)
        out[(size_t)(mbase + mrow0 + v) * C_SZ + c] = acc[j][v];
    }
  }
}

extern "C" void kernel_launch(void* const* d_in, const int* in_sizes, int n_in,
                              void* d_out, int out_size, void* d_ws, size_t ws_size,
                              hipStream_t stream) {
  const float* x  = (const float*)d_in[0];   // [4096][512]
  const float* sw = (const float*)d_in[1];   // [64][63][512]
  const float* sb = (const float*)d_in[2];   // [64][63]
  const float* ll = (const float*)d_in[3];   // [64][64][100]
  const float* tw = (const float*)d_in[4];   // [64]
  float* out = (float*)d_out;                // [4096][100]

  // workspace layout (≈41 MiB):
  char* ws = (char*)d_ws;
  __bf16* xb   = (__bf16*)(ws);                        //  4 MiB  x bf16
  __bf16* wp   = (__bf16*)(ws + (4ull  << 20));        //  4 MiB  packed weights
  __bf16* a2   = (__bf16*)(ws + (8ull  << 20));        // 32 MiB  leaf probs
  __bf16* dist = (__bf16*)(ws + (40ull << 20));        // <1 MiB  scaled leaf dists

  cvt_x_kernel  <<<(B_SZ * D_SZ) / 256, 256, 0, stream>>>(x, xb);
  pack_w_kernel <<<(K2   * D_SZ) / 256, 256, 0, stream>>>(sw, wp);
  dist_kernel   <<<K2 / 256,            256, 0, stream>>>(ll, tw, dist);
  gemm1_leafprob_kernel<<<(256 * 64) / 8, 256, 0, stream>>>(xb, wp, sb, a2);
  gemm2_kernel  <<<256 / 8,             256, 0, stream>>>(a2, dist, out);
}
